// Nequix_28836410425841
// MI455X (gfx1250) — matrix operand order repaired and checked by reading, not compiled, not run on hardware
//
#include <hip/hip_runtime.h>
#include <hip/hip_bf16.h>
#include <math.h>

typedef __attribute__((ext_vector_type(16))) _Float16 v16h;
typedef __attribute__((ext_vector_type(8)))  _Float16 v8h;
typedef __attribute__((ext_vector_type(8)))  float    v8f;

#define N_NODES 20000
#define N_EDGES 320000
#define WV 8              // waves per block in edge kernels
#define SQRT3_F 1.7320508075688772f
#define INV_SQRT3_F 0.5773502691896258f
#define PI_F 3.14159265358979323846f

// ---------- helpers ----------

// fast SiLU using the hardware reciprocal (v_rcp_f32, ~1 ulp): activation
// precision is dominated by f16 storage anyway. Avoids the IEEE
// div_scale/rcp/NR-refine division sequence (~12 VALU -> ~5 VALU).
__device__ __forceinline__ float silu_f(float x) {
    return x * __builtin_amdgcn_rcpf(1.0f + __expf(-x));
}

__device__ __forceinline__ v8f zero8() {
    v8f z = {0.f,0.f,0.f,0.f,0.f,0.f,0.f,0.f};
    return z;
}

__device__ __forceinline__ v8f wmma32(v16h a, v16h b, v8f c) {
    // D = A(16x32 f16) x B(32x16 f16) + C(16x16 f32)
    return __builtin_amdgcn_wmma_f32_16x16x32_f16(false, a, false, b, (short)0, c, false, false);
}

// A fragment: X row-major f16 in LDS, row stride `stride` halfs, K-block kb.
// Layout: lanes 0-15 row=lane, K kb..kb+15;
//         lanes 16-31 row=lane-16, K kb+16..kb+31.
__device__ __forceinline__ v16h load_A(const _Float16* X, int lane, int kb, int stride) {
    int m = lane & 15;
    int koff = (lane < 16) ? 0 : 16;
    const _Float16* p = X + m * stride + kb + koff;
    v8h lo = *(const v8h*)(p);
    v8h hi = *(const v8h*)(p + 8);
    v16h a;
#pragma unroll
    for (int i = 0; i < 8; ++i) { a[i] = lo[i]; a[i + 8] = hi[i]; }
    return a;
}

// B fragment: Wt packed as [n][kstride] f16 (K-contiguous per output column n).
// Layout (contiguous k, same as load_A): lanes 0-15 col=n0+lane, K kb+0..15; lanes 16-31 col, K kb+16..31.
__device__ __forceinline__ v16h load_B(const _Float16* Wt, int lane, int n0, int kb, int kstride) {
    int n = n0 + (lane & 15);
    int koff = (lane < 16) ? 0 : 16;
    return *(const v16h*)(Wt + n * kstride + kb + koff);   // 32B aligned
}

// store D tile with SiLU back to X (16x64 f16, row-major)
__device__ __forceinline__ void store_silu_tile(_Float16* X, int lane, int n0, v8f d) {
    int n = n0 + (lane & 15);
    int mb = (lane >> 4) << 3;
#pragma unroll
    for (int r = 0; r < 8; ++r) {
        float v = d[r];
        X[(mb + r) * 64 + n] = (_Float16)silu_f(v);
    }
}

// shared radial hidden stack: rb(16x8 padded to 32) -> 64 -> 64 -> 64 (SiLU each).
// Leaves h2 in Xa and returns its two A fragments.
__device__ __forceinline__ void radial_hidden(_Float16* Xa,
                                              const _Float16* wt0, const _Float16* wt1,
                                              const _Float16* wt2, int lane,
                                              v16h& a0, v16h& a1) {
    // layer 0: K=32 (8 real + 24 zero-pad), out 64
    v16h a = load_A(Xa, lane, 0, 64);
    v8f d0 = zero8(), d1 = zero8(), d2 = zero8(), d3 = zero8();
    d0 = wmma32(a, load_B(wt0, lane,  0, 0, 32), d0);
    d1 = wmma32(a, load_B(wt0, lane, 16, 0, 32), d1);
    d2 = wmma32(a, load_B(wt0, lane, 32, 0, 32), d2);
    d3 = wmma32(a, load_B(wt0, lane, 48, 0, 32), d3);
    __syncthreads();
    store_silu_tile(Xa, lane,  0, d0);
    store_silu_tile(Xa, lane, 16, d1);
    store_silu_tile(Xa, lane, 32, d2);
    store_silu_tile(Xa, lane, 48, d3);
    __syncthreads();
    // layer 1: K=64
    a0 = load_A(Xa, lane, 0, 64);
    a1 = load_A(Xa, lane, 32, 64);
    d0 = zero8(); d1 = zero8(); d2 = zero8(); d3 = zero8();
    d0 = wmma32(a0, load_B(wt1, lane,  0,  0, 64), d0);
    d0 = wmma32(a1, load_B(wt1, lane,  0, 32, 64), d0);
    d1 = wmma32(a0, load_B(wt1, lane, 16,  0, 64), d1);
    d1 = wmma32(a1, load_B(wt1, lane, 16, 32, 64), d1);
    d2 = wmma32(a0, load_B(wt1, lane, 32,  0, 64), d2);
    d2 = wmma32(a1, load_B(wt1, lane, 32, 32, 64), d2);
    d3 = wmma32(a0, load_B(wt1, lane, 48,  0, 64), d3);
    d3 = wmma32(a1, load_B(wt1, lane, 48, 32, 64), d3);
    __syncthreads();
    store_silu_tile(Xa, lane,  0, d0);
    store_silu_tile(Xa, lane, 16, d1);
    store_silu_tile(Xa, lane, 32, d2);
    store_silu_tile(Xa, lane, 48, d3);
    __syncthreads();
    // layer 2: K=64
    a0 = load_A(Xa, lane, 0, 64);
    a1 = load_A(Xa, lane, 32, 64);
    d0 = zero8(); d1 = zero8(); d2 = zero8(); d3 = zero8();
    d0 = wmma32(a0, load_B(wt2, lane,  0,  0, 64), d0);
    d0 = wmma32(a1, load_B(wt2, lane,  0, 32, 64), d0);
    d1 = wmma32(a0, load_B(wt2, lane, 16,  0, 64), d1);
    d1 = wmma32(a1, load_B(wt2, lane, 16, 32, 64), d1);
    d2 = wmma32(a0, load_B(wt2, lane, 32,  0, 64), d2);
    d2 = wmma32(a1, load_B(wt2, lane, 32, 32, 64), d2);
    d3 = wmma32(a0, load_B(wt2, lane, 48,  0, 64), d3);
    d3 = wmma32(a1, load_B(wt2, lane, 48, 32, 64), d3);
    __syncthreads();
    store_silu_tile(Xa, lane,  0, d0);
    store_silu_tile(Xa, lane, 16, d1);
    store_silu_tile(Xa, lane, 32, d2);
    store_silu_tile(Xa, lane, 48, d3);
    __syncthreads();
    a0 = load_A(Xa, lane, 0, 64);
    a1 = load_A(Xa, lane, 32, 64);
}

// edge preamble: lanes 0..15 compute rn, cutoff*bessel -> Xa (cols 0..7, zeros 8..31),
// Y1 -> sY, sender/receiver -> meta
__device__ __forceinline__ void edge_preamble(const float* __restrict__ disp,
                                              const int* __restrict__ senders,
                                              const int* __restrict__ receivers,
                                              int base, int lane,
                                              _Float16* Xa, float* sY, int* sSnd, int* sRcv) {
    if (lane < 16) {
        int e = base + lane;
        float dx = disp[e * 3 + 0];
        float dy = disp[e * 3 + 1];
        float dz = disp[e * 3 + 2];
        float sq = dx * dx + dy * dy + dz * dz;
        float rn = sqrtf(sq);
        float inv = (rn > 0.f) ? __builtin_amdgcn_rcpf(rn) : 1.f;
        sY[lane * 3 + 0] = SQRT3_F * dx * inv;
        sY[lane * 3 + 1] = SQRT3_F * dy * inv;
        sY[lane * 3 + 2] = SQRT3_F * dz * inv;
        sSnd[lane] = senders[e];
        sRcv[lane] = receivers[e];
        float u = rn * 0.2f;  // rn / CUTOFF
        float fc = (u < 1.f) ? (1.f + u * u * (-6.f + u * (8.f - 3.f * u))) : 0.f;
#pragma unroll
        for (int j = 0; j < 8; ++j) {
            float w = (float)(j + 1) * PI_F;
            float val = (rn > 0.f) ? (0.4f * __sinf(w * rn * 0.2f) * inv)
                                   : (0.4f * w * 0.2f);
            Xa[lane * 64 + j] = (_Float16)(val * fc);
        }
#pragma unroll
        for (int j = 8; j < 32; ++j) Xa[lane * 64 + j] = (_Float16)0.f;
    }
}

// ---------- kernels ----------

// convert radial weights fp32 -> f16, K-contiguous transposed (+ zero pad K for layer0)
__global__ void k_prep(const float* __restrict__ a10, const float* __restrict__ a11,
                       const float* __restrict__ a12, const float* __restrict__ a13,
                       const float* __restrict__ a20, const float* __restrict__ a21,
                       const float* __restrict__ a22, const float* __restrict__ a23,
                       _Float16* __restrict__ W) {
    int i = blockIdx.x * 256 + threadIdx.x;
    if (i >= 37888) return;
    if (i < 2048) {                       // l1_rw0: (8,64) -> [64n][32k] zero-pad
        int n = i / 32, k = i % 32;
        W[i] = (k < 8) ? (_Float16)a10[k * 64 + n] : (_Float16)0.f;
    } else if (i < 6144) {                // l1_rw1: (64,64) -> [64n][64k]
        int j = i - 2048; int n = j / 64, k = j % 64;
        W[i] = (_Float16)a11[k * 64 + n];
    } else if (i < 10240) {               // l1_rw2
        int j = i - 6144; int n = j / 64, k = j % 64;
        W[i] = (_Float16)a12[k * 64 + n];
    } else if (i < 11264) {               // l1_rw3: (64,16) -> [16n][64k]
        int j = i - 10240; int n = j / 64, k = j % 64;
        W[i] = (_Float16)a13[k * 16 + n];
    } else if (i < 13312) {               // l2_rw0
        int j = i - 11264; int n = j / 32, k = j % 32;
        W[i] = (k < 8) ? (_Float16)a20[k * 64 + n] : (_Float16)0.f;
    } else if (i < 17408) {               // l2_rw1
        int j = i - 13312; int n = j / 64, k = j % 64;
        W[i] = (_Float16)a21[k * 64 + n];
    } else if (i < 21504) {               // l2_rw2
        int j = i - 17408; int n = j / 64, k = j % 64;
        W[i] = (_Float16)a22[k * 64 + n];
    } else {                              // l2_rw3: (64,256) -> [256n][64k]
        int j = i - 21504; int n = j / 64, k = j % 64;
        W[i] = (_Float16)a23[k * 256 + n];
    }
}

__global__ void k_zero(float* __restrict__ p, int n) {
    int i = blockIdx.x * 256 + threadIdx.x;
    if (i < n) p[i] = 0.f;
}

// edge pass 1: radial MLP (WMMA) -> R1(16), build messages, scatter agg_s/agg_v
__global__ void __launch_bounds__(256) k_edge1(
        const float* __restrict__ disp, const int* __restrict__ species,
        const int* __restrict__ senders, const int* __restrict__ receivers,
        const float* __restrict__ lin1, const _Float16* __restrict__ W,
        float* __restrict__ agg_s, float* __restrict__ agg_v) {
    __shared__ __align__(32) _Float16 sXa[WV][16 * 64];
    __shared__ float sY[WV][48];
    __shared__ int   sSnd[WV][16], sRcv[WV][16];
    __shared__ float sR[WV][256];

    int wave = threadIdx.x >> 5;
    int lane = threadIdx.x & 31;
    int tile = blockIdx.x * WV + wave;
    int base = tile * 16;

    edge_preamble(disp, senders, receivers, base, lane,
                  sXa[wave], sY[wave], sSnd[wave], sRcv[wave]);
    __syncthreads();

    v16h a0, a1;
    radial_hidden(sXa[wave], W + 0, W + 2048, W + 6144, lane, a0, a1);

    // layer 3: 64 -> 16 (no activation)
    v8f r1 = zero8();
    r1 = wmma32(a0, load_B(W + 10240, lane, 0,  0, 64), r1);
    r1 = wmma32(a1, load_B(W + 10240, lane, 0, 32, 64), r1);
    {
        int n = lane & 15;
        int mb = (lane >> 4) << 3;
#pragma unroll
        for (int r = 0; r < 8; ++r) sR[wave][(mb + r) * 16 + n] = r1[r];
    }
    __syncthreads();

    // message + scatter: lane handles edge (lane&15), channels half*4..half*4+3
    {
        int el = lane & 15;
        int half = lane >> 4;
        int snd = sSnd[wave][el];
        int rcv = sRcv[wave][el];
        int sp = species[snd];
        float y0 = sY[wave][el * 3 + 0];
        float y1v = sY[wave][el * 3 + 1];
        float y2 = sY[wave][el * 3 + 2];
#pragma unroll
        for (int cc = 0; cc < 4; ++cc) {
            int c = half * 4 + cc;
            float sv = lin1[sp * 8 + c];
            atomicAdd(&agg_s[rcv * 8 + c], sv * sR[wave][el * 16 + c]);
            float rv = sv * sR[wave][el * 16 + 8 + c];
            atomicAdd(&agg_v[(rcv * 8 + c) * 3 + 0], rv * y0);
            atomicAdd(&agg_v[(rcv * 8 + c) * 3 + 1], rv * y1v);
            atomicAdd(&agg_v[(rcv * 8 + c) * 3 + 2], rv * y2);
        }
    }
}

// node pass 1: mixing, gates, x0/x1, then lin1s / lin1v projections
__global__ void __launch_bounds__(128) k_node1(
        const float* __restrict__ agg_s, const float* __restrict__ agg_v,
        const int* __restrict__ species,
        const float* __restrict__ w2s, const float* __restrict__ w2v,
        const float* __restrict__ skip1,
        const float* __restrict__ lin1s, const float* __restrict__ lin1v,
        float* __restrict__ x0o, float* __restrict__ s2n, float* __restrict__ v2n) {
    int n = blockIdx.x;
    int d = threadIdx.x;
    __shared__ float sA[8], sV[24];
    __shared__ float sX0[128], sX1[128 * 3];
    if (d < 8)  sA[d] = agg_s[n * 8 + d] * 0.25f;     // * 1/sqrt(AVG_N)
    if (d < 24) sV[d] = agg_v[n * 24 + d] * 0.25f;
    __syncthreads();
    int sp = species[n];
    float lo = skip1[(sp * 8 + sp) * 256 + d];
    float hi = skip1[(sp * 8 + sp) * 256 + 128 + d];
    float v0 = 0.f, v1 = 0.f, v2 = 0.f;
#pragma unroll
    for (int c = 0; c < 8; ++c) {
        float a = sA[c];
        lo += a * w2s[c * 256 + d];
        hi += a * w2s[c * 256 + 128 + d];
        float wv = w2v[c * 128 + d];
        v0 += sV[c * 3 + 0] * wv;
        v1 += sV[c * 3 + 1] * wv;
        v2 += sV[c * 3 + 2] * wv;
    }
    float x0 = silu_f(lo);
    float g  = silu_f(hi);
    sX0[d] = x0;
    sX1[d * 3 + 0] = v0 * g;
    sX1[d * 3 + 1] = v1 * g;
    sX1[d * 3 + 2] = v2 * g;
    x0o[n * 128 + d] = x0;
    __syncthreads();
    float s2 = 0.f, u0 = 0.f, u1 = 0.f, u2 = 0.f;
    for (int c = 0; c < 128; ++c) {
        s2 += sX0[c] * lin1s[c * 128 + d];
        float wv = lin1v[c * 128 + d];
        u0 += sX1[c * 3 + 0] * wv;
        u1 += sX1[c * 3 + 1] * wv;
        u2 += sX1[c * 3 + 2] * wv;
    }
    s2n[n * 128 + d] = s2;
    v2n[(n * 128 + d) * 3 + 0] = u0;
    v2n[(n * 128 + d) * 3 + 1] = u1;
    v2n[(n * 128 + d) * 3 + 2] = u2;
}

// edge pass 2: recompute radial (WMMA, 64->256 head), tensor-product messages,
// scatter agg2 (N x 256)
__global__ void __launch_bounds__(256) k_edge2(
        const float* __restrict__ disp,
        const int* __restrict__ senders, const int* __restrict__ receivers,
        const _Float16* __restrict__ W,
        const float* __restrict__ s2n, const float* __restrict__ v2n,
        float* __restrict__ agg2) {
    __shared__ __align__(32) _Float16 sXa[WV][16 * 64];
    __shared__ float sY[WV][48];
    __shared__ int   sSnd[WV][16], sRcv[WV][16];

    int wave = threadIdx.x >> 5;
    int lane = threadIdx.x & 31;
    int tile = blockIdx.x * WV + wave;
    int base = tile * 16;

    edge_preamble(disp, senders, receivers, base, lane,
                  sXa[wave], sY[wave], sSnd[wave], sRcv[wave]);
    __syncthreads();

    v16h a0, a1;
    radial_hidden(sXa[wave], W + 11264, W + 13312, W + 17408, lane, a0, a1);

    const _Float16* wt3 = W + 21504;  // [256n][64k]
    int nl = lane & 15;
    int mb = (lane >> 4) << 3;
#pragma unroll
    for (int t = 0; t < 16; ++t) {
        v8f d = zero8();
        d = wmma32(a0, load_B(wt3, lane, t * 16,  0, 64), d);
        d = wmma32(a1, load_B(wt3, lane, t * 16, 32, 64), d);
        int n = t * 16 + nl;  // feature 0..255 (uniform branch per tile)
#pragma unroll
        for (int r = 0; r < 8; ++r) {
            int el = mb + r;
            int snd = sSnd[wave][el];
            int rcv = sRcv[wave][el];
            float tpv;
            if (n < 128) {
                tpv = s2n[snd * 128 + n];
            } else {
                int f = n - 128;
                const float* vp = v2n + (size_t)(snd * 128 + f) * 3;
                tpv = (vp[0] * sY[wave][el * 3 + 0] +
                       vp[1] * sY[wave][el * 3 + 1] +
                       vp[2] * sY[wave][el * 3 + 2]) * INV_SQRT3_F;
            }
            atomicAdd(&agg2[(size_t)rcv * 256 + n], tpv * d[r]);
        }
    }
}

// node pass 2: w2 + species skip, SiLU, readout dot, LDS reduce
__global__ void __launch_bounds__(128) k_node2(
        const float* __restrict__ agg2, const float* __restrict__ x0,
        const int* __restrict__ species,
        const float* __restrict__ w2, const float* __restrict__ skip2,
        const float* __restrict__ rw, float* __restrict__ out) {
    int n = blockIdx.x;
    int d = threadIdx.x;
    __shared__ float sG[256], sX[128], sRed[128];
    sG[d]       = agg2[n * 256 + d] * 0.25f;
    sG[d + 128] = agg2[n * 256 + 128 + d] * 0.25f;
    sX[d] = x0[n * 128 + d];
    __syncthreads();
    int sp = species[n];
    float acc = 0.f;
    for (int k = 0; k < 256; ++k) acc += sG[k] * w2[k * 128 + d];
    for (int c = 0; c < 128; ++c) acc += sX[c] * skip2[(sp * 128 + c) * 128 + d];
    float feat = silu_f(acc);
    sRed[d] = feat * rw[d];
    __syncthreads();
#pragma unroll
    for (int s = 64; s > 0; s >>= 1) {
        if (d < s) sRed[d] += sRed[d + s];
        __syncthreads();
    }
    if (d == 0) out[n] = sRed[0];
}

// ---------- launcher ----------

extern "C" void kernel_launch(void* const* d_in, const int* in_sizes, int n_in,
                              void* d_out, int out_size, void* d_ws, size_t ws_size,
                              hipStream_t stream) {
    (void)in_sizes; (void)n_in; (void)out_size; (void)ws_size;
    const float* disp      = (const float*)d_in[0];
    const int*   species   = (const int*)d_in[1];
    const int*   senders   = (const int*)d_in[2];
    const int*   receivers = (const int*)d_in[3];
    const float* l1_lin1   = (const float*)d_in[4];
    const float* l1_rw0    = (const float*)d_in[5];
    const float* l1_rw1    = (const float*)d_in[6];
    const float* l1_rw2    = (const float*)d_in[7];
    const float* l1_rw3    = (const float*)d_in[8];
    const float* l1_w2s    = (const float*)d_in[9];
    const float* l1_w2v    = (const float*)d_in[10];
    const float* l1_skip   = (const float*)d_in[11];
    const float* l2_lin1s  = (const float*)d_in[12];
    const float* l2_lin1v  = (const float*)d_in[13];
    const float* l2_rw0    = (const float*)d_in[14];
    const float* l2_rw1    = (const float*)d_in[15];
    const float* l2_rw2    = (const float*)d_in[16];
    const float* l2_rw3    = (const float*)d_in[17];
    const float* l2_w2     = (const float*)d_in[18];
    const float* l2_skip   = (const float*)d_in[19];
    const float* readout_w = (const float*)d_in[20];

    char* ws = (char*)d_ws;
    _Float16* W     = (_Float16*)(ws);                // 75776 B (f16 weights)
    float*    agg_s = (float*)(ws + 81920);           // N*8
    float*    agg_v = (float*)(ws + 721920);          // N*24
    float*    agg2  = (float*)(ws + 2641920);         // N*256
    float*    x0    = (float*)(ws + 23121920);        // N*128
    float*    s2n   = (float*)(ws + 33361920);        // N*128
    float*    v2n   = (float*)(ws + 43601920);        // N*128*3

    k_prep<<<148, 256, 0, stream>>>(l1_rw0, l1_rw1, l1_rw2, l1_rw3,
                                    l2_rw0, l2_rw1, l2_rw2, l2_rw3, W);
    // agg_s, agg_v, agg2 are contiguous: N*288 floats
    k_zero<<<(N_NODES * 288 + 255) / 256, 256, 0, stream>>>(agg_s, N_NODES * 288);
    k_edge1<<<N_EDGES / (16 * WV), 256, 0, stream>>>(disp, species, senders, receivers,
                                                     l1_lin1, W, agg_s, agg_v);
    k_node1<<<N_NODES, 128, 0, stream>>>(agg_s, agg_v, species, l1_w2s, l1_w2v, l1_skip,
                                         l2_lin1s, l2_lin1v, x0, s2n, v2n);
    k_edge2<<<N_EDGES / (16 * WV), 256, 0, stream>>>(disp, senders, receivers, W,
                                                     s2n, v2n, agg2);
    k_node2<<<N_NODES, 128, 0, stream>>>(agg2, x0, species, l2_w2, l2_skip,
                                         readout_w, (float*)d_out);
}
